// Event_RGB_Dynamic_Fusion_Module_22771916603485
// MI455X (gfx1250) — matrix operand order repaired and checked by reading, not compiled, not run on hardware
//
#include <hip/hip_runtime.h>

typedef __attribute__((ext_vector_type(16))) _Float16 v16h;
typedef __attribute__((ext_vector_type(8)))  float    v8f;
typedef __attribute__((ext_vector_type(4)))  unsigned int u32x4;
typedef __attribute__((ext_vector_type(8)))  int      i32x8;
typedef __attribute__((ext_vector_type(4)))  int      i32x4;

#define C_RGB 64
#define C_EV  32
#define CIN   96
#define MID   32
#define HH    256
#define WW    256
#define NTAP  9
#define OUTC  (C_RGB * NTAP)   // 576
#define PW    258              // padded row width (reflect pad 1)

// ---- dynamic LDS layout (bytes) ----
#define OFF_PATCH 0
#define SZ_PATCH  (3 * C_RGB * PW * 4)        // 198144 f32 [ri][c][xx]
#define OFF_EV    (OFF_PATCH + SZ_PATCH)
#define SZ_EV     (C_EV * WW * 4)             // 32768  f32 [c][x]
#define OFF_W1    (OFF_EV + SZ_EV)
#define SZ_W1     (MID * CIN * 2)             // 6144   f16 [m][c]
#define OFF_W2    (OFF_W1 + SZ_W1)
#define SZ_W2     (OUTC * MID * 2)            // 36864  f16 [o][m]
#define OFF_H     (OFF_W2 + SZ_W2)
#define SZ_H      (WW * MID * 2)              // 16384  f16 [px][m]
#define OFF_B1    (OFF_H + SZ_H)
#define SZ_B1     (MID * 4)
#define OFF_B2    (OFF_B1 + SZ_B1)
#define SZ_B2     (OUTC * 4)
#define SMEM_BYTES (OFF_B2 + SZ_B2)           // 292736

// K index held by (vgpr v, half h) for a 16-bit A operand (16x32), ISA 7.12.2
__device__ __forceinline__ int a_k(int v, int h, int hi16) {
  int base = (v < 4) ? (2 * v + h) : (16 + 2 * (v - 4) + h);
  return base + (hi16 ? 8 : 0);
}
// K index held by (vgpr v, half h) for a 16-bit B operand (32x16)
__device__ __forceinline__ int b_k(int v, int h, int hi16) {
  return a_k(v, h, hi16);
}

__device__ __forceinline__ unsigned lds_off(const void* p) {
  return (unsigned)(unsigned long long)(uintptr_t)p;
}

// TDM 2D tile load: tile_d0 elements (f32, contiguous) x tile_d1 lines
// (line stride d0_stride elems); optional LDS pad of 2 DWORDs per 256 DWORDs.
__device__ __forceinline__ void tdm_load_2d(unsigned lds_addr, unsigned long long ga,
                                            unsigned tile_d0, unsigned tile_d1,
                                            unsigned long long d0_stride, int pad) {
  u32x4 g0;
  g0[0] = 1u;                                  // count=1, user descriptor
  g0[1] = lds_addr;                            // lds_addr [63:32]
  g0[2] = (unsigned)(ga & 0xFFFFFFFFull);      // global_addr[31:0]
  g0[3] = (unsigned)(ga >> 32) | (2u << 30);   // global_addr[56:32] | type=2
  i32x8 g1;
  // data_size=2 (4B) [17:16] | pad_enable [20] | pad_interval=7 (256 DW) [24:22]
  // | pad_amount=1 (2 DW) [31:25]
  g1[0] = (2 << 16) | (pad ? ((1 << 20) | (7 << 22) | (1 << 25)) : 0);
  g1[1] = (int)(tile_d0 << 16);                // tensor_dim0[15:0] at [79:64]... bits[79:48]
  g1[2] = (int)(tile_d1 << 16);                // tensor_dim0[31:16]=0, tensor_dim1[15:0]
  g1[3] = (int)(tile_d0 << 16);                // tensor_dim1[31:16]=0, tile_dim0
  g1[4] = (int)tile_d1;                        // tile_dim1, tile_dim2=0
  g1[5] = (int)(unsigned)(d0_stride & 0xFFFFFFFFull);   // tensor_dim0_stride[31:0]
  g1[6] = (int)(unsigned)(d0_stride >> 32);    // stride[47:32]; dim1_stride[15:0]=0
  g1[7] = 0;
  i32x4 z4 = {0, 0, 0, 0};
#if defined(__clang_major__) && (__clang_major__ >= 23)
  i32x8 z8 = {0, 0, 0, 0, 0, 0, 0, 0};
  __builtin_amdgcn_tensor_load_to_lds(g0, g1, z4, z4, z8, 0);
#else
  __builtin_amdgcn_tensor_load_to_lds(g0, g1, z4, z4, 0);
#endif
}

__global__ __launch_bounds__(256)
void fusion_kernel(const float* __restrict__ rgb, const float* __restrict__ ev,
                   const float* __restrict__ W1, const float* __restrict__ b1,
                   const float* __restrict__ W2, const float* __restrict__ b2,
                   float* __restrict__ out)
{
  extern __shared__ char smem[];
  float*    sPatch = (float*)(smem + OFF_PATCH);
  float*    sEv    = (float*)(smem + OFF_EV);
  _Float16* sW1    = (_Float16*)(smem + OFF_W1);
  _Float16* sW2    = (_Float16*)(smem + OFF_W2);
  _Float16* sH     = (_Float16*)(smem + OFF_H);
  float*    sB1    = (float*)(smem + OFF_B1);
  float*    sB2    = (float*)(smem + OFF_B2);

  const int tid  = threadIdx.x;
  const int lane = tid & 31;
  const int wave = tid >> 5;
  const int hi16 = (lane & 16) ? 1 : 0;
  const int ln   = lane & 15;
  const int blk  = blockIdx.x;
  const int b    = blk / HH;
  const int y    = blk % HH;

  // ---- Tensor Data Mover: stage 3 reflected rgb rows ([c][258] slabs with LDS
  // row padding) + the event row, issued by wave 0, tracked via TENSORcnt ----
  if (wave == 0) {
#pragma unroll
    for (int ri = 0; ri < 3; ++ri) {
      int gy = y + ri - 1;
      gy = (gy < 0) ? -gy : ((gy >= HH) ? (2 * HH - 2 - gy) : gy);
      unsigned long long ga =
          (unsigned long long)(uintptr_t)(rgb + (((size_t)b * C_RGB) * HH + gy) * WW);
      tdm_load_2d(lds_off(sPatch + ri * C_RGB * PW + 1), ga,
                  WW, C_RGB, (unsigned long long)HH * WW, /*pad=*/1);
    }
    unsigned long long gae =
        (unsigned long long)(uintptr_t)(ev + (((size_t)b * C_EV) * HH + y) * WW);
    tdm_load_2d(lds_off(sEv), gae, WW, C_EV, (unsigned long long)HH * WW, /*pad=*/0);
    __builtin_amdgcn_s_wait_tensorcnt(0);
  }

  // ---- reflect halo columns (xx=0 and xx=257) + weights/biases, all waves ----
  for (int i = tid; i < 3 * C_RGB; i += 256) {
    int c = i % C_RGB, ri = i / C_RGB;
    int gy = y + ri - 1;
    gy = (gy < 0) ? -gy : ((gy >= HH) ? (2 * HH - 2 - gy) : gy);
    const float* rowp = rgb + (((size_t)b * C_RGB + c) * HH + gy) * WW;
    sPatch[(ri * C_RGB + c) * PW + 0]      = rowp[1];       // x=-1 -> 1
    sPatch[(ri * C_RGB + c) * PW + PW - 1] = rowp[WW - 2];  // x=256 -> 254
  }
  for (int i = tid; i < MID * CIN; i += 256)  sW1[i] = (_Float16)W1[i];
  for (int i = tid; i < OUTC * MID; i += 256) sW2[i] = (_Float16)W2[i];
  if (tid < MID) sB1[tid] = b1[tid];
  for (int i = tid; i < OUTC; i += 256) sB2[i] = b2[i];
  __syncthreads();

  // ===== Phase 1: h = relu(W1 * fused + b1), D = W1_block(16m x 96k) x fused^T(96k x 16px) =====
  for (int pt = 0; pt < 2; ++pt) {
    const int x0 = wave * 32 + pt * 16;
    const int x  = x0 + ln;
    // B operand: B[k=c, n=px] = fused[b, c, y, x]; all channels now in LDS
    v16h Bf[3];
#pragma unroll
    for (int kc = 0; kc < 3; ++kc) {
#pragma unroll
      for (int e = 0; e < 16; ++e) {
        int v = e >> 1, h = e & 1;
        int c = kc * 32 + b_k(v, h, hi16);
        float val = (c < C_RGB)
            ? sPatch[(1 * C_RGB + c) * PW + x + 1]   // rgb row y
            : sEv[(c - C_RGB) * WW + x];
        Bf[kc][e] = (_Float16)val;
      }
    }
#pragma unroll
    for (int rt = 0; rt < 2; ++rt) {
      v8f d = {};
#pragma unroll
      for (int kc = 0; kc < 3; ++kc) {
        v16h A;
#pragma unroll
        for (int e = 0; e < 16; ++e) {
          int v = e >> 1, h = e & 1;
          int m = rt * 16 + ln;
          int c = kc * 32 + a_k(v, h, hi16);
          A[e] = sW1[m * CIN + c];
        }
        d = __builtin_amdgcn_wmma_f32_16x16x32_f16(false, A, false, Bf[kc],
                                                   (short)0, d, false, false);
      }
      // relu + bias, restage h^T as f16: sH[px][m]
#pragma unroll
      for (int r = 0; r < 8; ++r) {
        int m = rt * 16 + r + (hi16 ? 8 : 0);
        float hv = fmaxf(d[r] + sB1[m], 0.0f);
        sH[x * MID + m] = (_Float16)hv;
      }
    }
  }
  __syncthreads();

  // ===== Phase 2: dk = W2*h + b2 fused with dynamic-kernel apply =====
  // D = W2_block(16c x 32m) x h^T(32m x 16px): rows=channels, cols=pixels
  for (int pt = 0; pt < 2; ++pt) {
    const int x0 = wave * 32 + pt * 16;
    const int x  = x0 + ln;
    v16h Bh;
#pragma unroll
    for (int e = 0; e < 16; ++e) {
      int v = e >> 1, h = e & 1;
      Bh[e] = sH[x * MID + b_k(v, h, hi16)];
    }
#pragma unroll
    for (int nt = 0; nt < 4; ++nt) {
      float acc[8];
#pragma unroll
      for (int r = 0; r < 8; ++r) acc[r] = 0.0f;
      const int cl = nt * 16 + ln;            // A-row channel for this lane
#pragma unroll
      for (int tap = 0; tap < NTAP; ++tap) {
        const int o = cl * NTAP + tap;
        v16h A;
#pragma unroll
        for (int e = 0; e < 16; ++e) {
          int v = e >> 1, h = e & 1;
          A[e] = sW2[o * MID + a_k(v, h, hi16)];
        }
        v8f zero = {};
        v8f d = __builtin_amdgcn_wmma_f32_16x16x32_f16(false, A, false, Bh,
                                                       (short)0, zero, false, false);
        const int ri = tap / 3, dj = tap % 3;   // patch xx = x + dj
#pragma unroll
        for (int r = 0; r < 8; ++r) {
          int c = nt * 16 + r + (hi16 ? 8 : 0);
          float pv  = sPatch[(ri * C_RGB + c) * PW + x + dj];
          float dkv = d[r] + sB2[c * NTAP + tap];
          acc[r] += pv * dkv;
        }
      }
#pragma unroll
      for (int r = 0; r < 8; ++r) {
        int c = nt * 16 + r + (hi16 ? 8 : 0);
        out[(((size_t)b * C_RGB + c) * HH + y) * WW + x] = acc[r];  // coalesced in x
      }
    }
  }
}

extern "C" void kernel_launch(void* const* d_in, const int* in_sizes, int n_in,
                              void* d_out, int out_size, void* d_ws, size_t ws_size,
                              hipStream_t stream) {
  const float* rgb = (const float*)d_in[0];
  const float* ev  = (const float*)d_in[1];
  const float* W1  = (const float*)d_in[2];
  const float* b1  = (const float*)d_in[3];
  const float* W2  = (const float*)d_in[4];
  const float* b2  = (const float*)d_in[5];
  float* out = (float*)d_out;

  (void)hipFuncSetAttribute((const void*)fusion_kernel,
                            hipFuncAttributeMaxDynamicSharedMemorySize, SMEM_BYTES);

  dim3 grid(4 * HH);   // one workgroup per (b, y) row
  dim3 block(256);     // 8 wave32
  fusion_kernel<<<grid, block, SMEM_BYTES, stream>>>(rgb, ev, W1, b1, W2, b2, out);
}
